// BoundarySubBranch_50148038148688
// MI455X (gfx1250) — matrix operand order repaired and checked by reading, not compiled, not run on hardware
//
#include <hip/hip_runtime.h>
#include <hip/hip_bf16.h>

typedef unsigned short ushort_t;
typedef unsigned int uint_t;
typedef __attribute__((ext_vector_type(16))) __bf16   v16bf;
typedef __attribute__((ext_vector_type(8)))  float    v8f;
typedef __attribute__((ext_vector_type(8)))  ushort_t u16x8;

union FragU { u16x8 h[2]; v16bf bf; };

#define N_B   8
#define C_CH  512
#define T_LEN 4096
#define KW    3
#define KDIM  (C_CH * KW)       // 1536
#define NGRP  32
#define CPG   (C_CH / NGRP)     // 16 channels per group
#define EPSV  1e-5f

// workspace layout (bytes)
#define Y_OFF   0u                         // conv1 output, fp32: 8*512*4096*4 = 64 MB
#define HT_OFF  67108864u                  // hT[n][t][c] bf16: 32 MB
#define ST_OFF  100663296u                 // stats (mu,rsig) per (n,group): 2 KB
#define W1_OFF  100665344u                 // conv_w bf16 [cout][tap*512+cin]: 1.5 MB
#define W2_OFF  102238208u                 // dcn_w  bf16 same layout: 1.5 MB

#define OFFSET_ELEMS (N_B * 2 * T_LEN)     // 65536 floats, then offset_feat

#define NTILE 128                          // t-columns per block (8 subtiles/wave)
#define NSUB  (NTILE / 16)

__device__ __forceinline__ ushort_t f2bf(float f) {
  uint_t u = __float_as_uint(f);
  uint_t r = u + 0x7FFFu + ((u >> 16) & 1u);   // round to nearest even
  return (ushort_t)(r >> 16);
}
__device__ __forceinline__ float bf2f(ushort_t s) {
  return __uint_as_float(((uint_t)s) << 16);
}

// ---------------------------------------------------------------------------
// K0: convert conv_w / dcn_w (O,I,3) fp32 -> bf16 rearranged [o][k*512 + i]
// ---------------------------------------------------------------------------
__global__ void k_wprep(const float* __restrict__ w1f, const float* __restrict__ w2f,
                        ushort_t* __restrict__ w1b, ushort_t* __restrict__ w2b) {
  int idx = blockIdx.x * blockDim.x + threadIdx.x;
  const int total = C_CH * C_CH * KW;      // 786432
  if (idx < total) {
    int k = idx % KW; int i = (idx / KW) % C_CH; int o = idx / (KW * C_CH);
    w1b[o * KDIM + k * C_CH + i] = f2bf(w1f[idx]);
  } else if (idx < 2 * total) {
    int j = idx - total;
    int k = j % KW; int i = (j / KW) % C_CH; int o = j / (KW * C_CH);
    w2b[o * KDIM + k * C_CH + i] = f2bf(w2f[j]);
  }
}

// ---------------------------------------------------------------------------
// WMMA tile engine: block = 256 threads = 8 waves.
// Block tile: 128 cout x 128 t. Wave w -> cout rows [base + 16w, +16), 8 accums.
// A fragment (16-bit A 16x32 layout): lane l: m=l&15, hi=l>>4,
//   elems 0..7  = K (8*hi   + 0..7), elems 8..15 = K (16+8*hi + 0..7)
// B fragment (32x16): lane l: n=l&15, hi=l>>4, elems 0..7 = K (8*hi + 0..7), 8..15 = K (16+8*hi + 0..7)
// ---------------------------------------------------------------------------
#define BSTRIDE 40   // LDS row stride in halves; keeps all b128 loads 16B aligned

// K1: conv1d 512->512 K=3 pad=1, output y fp32 (bias added, no activation)
__global__ void __launch_bounds__(256)
k_conv1(const float* __restrict__ x, const ushort_t* __restrict__ wb,
        const float* __restrict__ bias, float* __restrict__ y) {
  __shared__ alignas(16) ushort_t sX[(NTILE + 2) * BSTRIDE];  // halo +-1, 32 c
  const int tbase = blockIdx.x * NTILE;
  const int cobase = blockIdx.y * 128;
  const int n = blockIdx.z;
  const int tid = threadIdx.x;
  const int wv = tid >> 5, l = tid & 31;
  const int m = l & 15, hi = l >> 4;

  v8f acc[NSUB];
#pragma unroll
  for (int i = 0; i < NSUB; ++i) acc[i] = v8f{};

  for (int cc = 0; cc < 16; ++cc) {        // 16 chunks of 32 input channels
    const int cbase = cc * 32;
    __syncthreads();
    for (int idx = tid; idx < 32 * (NTILE + 2); idx += 256) {  // stage transposed bf16
      int c = idx / (NTILE + 2), t = idx % (NTILE + 2);
      int tg = tbase - 1 + t;
      float v = (tg >= 0 && tg < T_LEN)
                  ? x[(n * C_CH + cbase + c) * T_LEN + tg] : 0.0f;
      sX[t * BSTRIDE + c] = f2bf(v);
    }
    __syncthreads();

#pragma unroll
    for (int tap = 0; tap < KW; ++tap) {
      FragU fa;
      const ushort_t* wp = wb + (cobase + wv * 16 + m) * KDIM + tap * C_CH + cbase + hi * 8;
      fa.h[0] = *(const u16x8*)(wp);
      fa.h[1] = *(const u16x8*)(wp + 16);
#pragma unroll
      for (int ns = 0; ns < NSUB; ++ns) {
        FragU fb;
        const ushort_t* bp = &sX[((l & 15) + ns * 16 + tap) * BSTRIDE + 8 * hi];
        fb.h[0] = *(const u16x8*)(bp);
        fb.h[1] = *(const u16x8*)(bp + 16);
        acc[ns] = __builtin_amdgcn_wmma_f32_16x16x32_bf16(
            false, fa.bf, false, fb.bf, (short)0, acc[ns], false, false);
      }
    }
  }

  // epilogue: add bias, store fp32
#pragma unroll
  for (int ns = 0; ns < NSUB; ++ns) {
#pragma unroll
    for (int r = 0; r < 8; ++r) {
      int row = cobase + wv * 16 + r + 8 * hi;
      int t = tbase + ns * 16 + (l & 15);
      y[(n * C_CH + row) * T_LEN + t] = acc[ns][r] + bias[row];
    }
  }
}

// K2: group-norm statistics: one block per (group, n)
__global__ void __launch_bounds__(256)
k_gnstats(const float* __restrict__ y, float* __restrict__ st) {
  __shared__ float sSum[256], sSq[256];
  const int g = blockIdx.x, n = blockIdx.y, tid = threadIdx.x;
  const float* base = y + (n * C_CH + g * CPG) * T_LEN;
  float s = 0.f, q = 0.f;
  for (int r = 0; r < CPG; ++r)
    for (int i = tid; i < T_LEN; i += 256) {
      float v = base[r * T_LEN + i];
      s += v; q += v * v;
    }
  sSum[tid] = s; sSq[tid] = q;
  __syncthreads();
  for (int off = 128; off > 0; off >>= 1) {
    if (tid < off) { sSum[tid] += sSum[tid + off]; sSq[tid] += sSq[tid + off]; }
    __syncthreads();
  }
  if (tid == 0) {
    const float inv = 1.0f / (float)(CPG * T_LEN);
    float mu = sSum[0] * inv;
    float var = sSq[0] * inv - mu * mu;
    st[(n * NGRP + g) * 2 + 0] = mu;
    st[(n * NGRP + g) * 2 + 1] = rsqrtf(var + EPSV);
  }
}

// K3: normalize + relu, write transposed bf16 hT[n][t][c]
__global__ void __launch_bounds__(256)
k_gnnorm(const float* __restrict__ y, const float* __restrict__ st,
         const float* __restrict__ gam, const float* __restrict__ bet,
         ushort_t* __restrict__ hT) {
  __shared__ ushort_t sT[32 * 33];
  const int tbase = blockIdx.x * 32, cbase = blockIdx.y * 32, n = blockIdx.z;
  const int tid = threadIdx.x;
#pragma unroll
  for (int p = 0; p < 4; ++p) {
    int c = p * 8 + (tid >> 5), t = tid & 31;
    int ch = cbase + c, g = ch >> 4;
    float mu = st[(n * NGRP + g) * 2 + 0];
    float rs = st[(n * NGRP + g) * 2 + 1];
    float v = y[(n * C_CH + ch) * T_LEN + tbase + t];
    v = fmaxf((v - mu) * rs * gam[ch] + bet[ch], 0.0f);
    sT[c * 33 + t] = f2bf(v);
  }
  __syncthreads();
#pragma unroll
  for (int p = 0; p < 4; ++p) {
    int t = p * 8 + (tid >> 5), c = tid & 31;
    hT[(n * T_LEN + tbase + t) * C_CH + cbase + c] = sT[c * 33 + t];
  }
}

// K4: deformable conv 512->512: sample hT (lerp, offsets shared over c),
// stage B tiles for all 3 taps in LDS, WMMA, relu -> offset_feat fp32
__global__ void __launch_bounds__(256)
k_dconv(const ushort_t* __restrict__ hT, const float* __restrict__ locs,
        const int* __restrict__ stridep, const ushort_t* __restrict__ wb,
        const float* __restrict__ bias, float* __restrict__ ofeat) {
  __shared__ alignas(16) ushort_t sB[3 * NTILE * BSTRIDE];
  __shared__ int   sI0[3 * NTILE];
  __shared__ float sW1[3 * NTILE];
  const int tbase = blockIdx.x * NTILE;
  const int cobase = blockIdx.y * 128;
  const int n = blockIdx.z;
  const int tid = threadIdx.x;
  const int wv = tid >> 5, l = tid & 31;
  const int m = l & 15, hi = l >> 4;

  {  // per-(tap,t) sampling positions
    const float inv_s = 1.0f / (float)(*stridep);
    for (int idx = tid; idx < 3 * NTILE; idx += 256) {
      int k = idx / NTILE, t = idx % NTILE;
      int tg = tbase + t;
      float off = 0.0f;
      if (k == 0) off = -locs[(n * 2 + 0) * T_LEN + tg] * inv_s;
      if (k == 2) off =  locs[(n * 2 + 1) * T_LEN + tg] * inv_s;
      float pos = (float)tg + (float)(k - 1) + off;
      float p0 = floorf(pos);
      sI0[idx] = (int)p0;
      sW1[idx] = pos - p0;
    }
  }

  v8f acc[NSUB];
#pragma unroll
  for (int i = 0; i < NSUB; ++i) acc[i] = v8f{};

  for (int cc = 0; cc < 16; ++cc) {
    const int cbase = cc * 32;
    __syncthreads();
    for (int idx = tid; idx < 3 * NTILE * 32; idx += 256) {   // lerp-sampled stage
      int c = idx & 31, rt = idx >> 5;         // rt = tap*NTILE + t
      int i0 = sI0[rt], i1 = i0 + 1;
      float w1 = sW1[rt];
      float v0 = (i0 >= 0 && i0 < T_LEN)
                   ? bf2f(hT[(n * T_LEN + i0) * C_CH + cbase + c]) : 0.0f;
      float v1 = (i1 >= 0 && i1 < T_LEN)
                   ? bf2f(hT[(n * T_LEN + i1) * C_CH + cbase + c]) : 0.0f;
      sB[rt * BSTRIDE + c] = f2bf(v0 * (1.0f - w1) + v1 * w1);
    }
    __syncthreads();

#pragma unroll
    for (int tap = 0; tap < KW; ++tap) {
      FragU fa;
      const ushort_t* wp = wb + (cobase + wv * 16 + m) * KDIM + tap * C_CH + cbase + hi * 8;
      fa.h[0] = *(const u16x8*)(wp);
      fa.h[1] = *(const u16x8*)(wp + 16);
#pragma unroll
      for (int ns = 0; ns < NSUB; ++ns) {
        FragU fb;
        const ushort_t* bp = &sB[(tap * NTILE + (l & 15) + ns * 16) * BSTRIDE + 8 * hi];
        fb.h[0] = *(const u16x8*)(bp);
        fb.h[1] = *(const u16x8*)(bp + 16);
        acc[ns] = __builtin_amdgcn_wmma_f32_16x16x32_bf16(
            false, fa.bf, false, fb.bf, (short)0, acc[ns], false, false);
      }
    }
  }

#pragma unroll
  for (int ns = 0; ns < NSUB; ++ns) {
#pragma unroll
    for (int r = 0; r < 8; ++r) {
      int row = cobase + wv * 16 + r + 8 * hi;
      int t = tbase + ns * 16 + (l & 15);
      ofeat[(n * C_CH + row) * T_LEN + t] = fmaxf(acc[ns][r] + bias[row], 0.0f);
    }
  }
}

// K5: tiny output conv 512->2, K=3 pad=1, fp32 VALU (0.2 GFLOP)
__global__ void __launch_bounds__(256)
k_outconv(const float* __restrict__ ofeat, const float* __restrict__ w,
          const float* __restrict__ b, float* __restrict__ off) {
  __shared__ float sW[C_CH * KW];
  const int tbase = blockIdx.x * 256, o = blockIdx.y, n = blockIdx.z;
  const int tid = threadIdx.x;
  for (int i = tid; i < C_CH * KW; i += 256) sW[i] = w[o * C_CH * KW + i];
  __syncthreads();
  const int t = tbase + tid;
  float acc = b[o];
  for (int c = 0; c < C_CH; ++c) {
    const float* fr = ofeat + (n * C_CH + c) * T_LEN;
    float xm = (t - 1 >= 0)    ? fr[t - 1] : 0.0f;
    float x0 = fr[t];
    float xp = (t + 1 < T_LEN) ? fr[t + 1] : 0.0f;
    acc += sW[c * KW + 0] * xm + sW[c * KW + 1] * x0 + sW[c * KW + 2] * xp;
  }
  off[(n * 2 + o) * T_LEN + t] = acc;
}

extern "C" void kernel_launch(void* const* d_in, const int* in_sizes, int n_in,
                              void* d_out, int out_size, void* d_ws, size_t ws_size,
                              hipStream_t stream) {
  const float* feat   = (const float*)d_in[0];
  const float* locs   = (const float*)d_in[1];
  const float* conv_w = (const float*)d_in[2];
  const float* conv_b = (const float*)d_in[3];
  const float* gn_g   = (const float*)d_in[4];
  const float* gn_b   = (const float*)d_in[5];
  const float* dcn_w  = (const float*)d_in[6];
  const float* dcn_b  = (const float*)d_in[7];
  const float* out_w  = (const float*)d_in[8];
  const float* out_b  = (const float*)d_in[9];
  const int*   stride = (const int*)d_in[10];

  char* ws = (char*)d_ws;
  float*    y   = (float*)(ws + Y_OFF);
  ushort_t* hT  = (ushort_t*)(ws + HT_OFF);
  float*    st  = (float*)(ws + ST_OFF);
  ushort_t* w1b = (ushort_t*)(ws + W1_OFF);
  ushort_t* w2b = (ushort_t*)(ws + W2_OFF);

  float* offset = (float*)d_out;                 // (8,2,4096)
  float* ofeat  = (float*)d_out + OFFSET_ELEMS;  // (8,512,4096)

  k_wprep<<<dim3((2 * C_CH * C_CH * KW + 255) / 256), 256, 0, stream>>>(
      conv_w, dcn_w, w1b, w2b);
  k_conv1<<<dim3(T_LEN / NTILE, C_CH / 128, N_B), 256, 0, stream>>>(
      feat, w1b, conv_b, y);
  k_gnstats<<<dim3(NGRP, N_B), 256, 0, stream>>>(y, st);
  k_gnnorm<<<dim3(T_LEN / 32, C_CH / 32, N_B), 256, 0, stream>>>(
      y, st, gn_g, gn_b, hT);
  k_dconv<<<dim3(T_LEN / NTILE, C_CH / 128, N_B), 256, 0, stream>>>(
      hT, locs, stride, w2b, dcn_b, ofeat);
  k_outconv<<<dim3(T_LEN / 256, 2, N_B), 256, 0, stream>>>(
      ofeat, out_w, out_b, offset);
}
